// GatedDeltaNetLayer_9079560864064
// MI455X (gfx1250) — matrix operand order repaired and checked by reading, not compiled, not run on hardware
//
#include <hip/hip_runtime.h>
#include <hip/hip_bf16.h>
#include <cstdint>
#include <cstddef>

// ---------------------------------------------------------------------------
// Problem constants (fixed by reference setup_inputs)
// ---------------------------------------------------------------------------
#define BATCH   4
#define TSEQ    2048
#define DMODEL  1024
#define NH      4
#define HKD     256      // per-head key dim
#define HVD     512      // per-head value dim
#define DKEY    1024
#define DVAL    2048
#define BT      (BATCH * TSEQ)   // 8192 rows

typedef __bf16 bf16;
typedef __attribute__((ext_vector_type(16))) __bf16 v16bf;
typedef __attribute__((ext_vector_type(8)))  __bf16 v8bf;
typedef __attribute__((ext_vector_type(8)))  float  v8f;

__device__ __forceinline__ float sigmoidf_(float x) {
    return 1.0f / (1.0f + __expf(-x));
}

// ---------------------------------------------------------------------------
// 1) LayerNorm over D=1024; emits f32 + bf16 copies (bf16 feeds WMMA GEMMs)
//    grid = BT blocks, 256 threads (4 elems/thread)
// ---------------------------------------------------------------------------
__global__ __launch_bounds__(256) void layernorm_kernel(
    const float* __restrict__ x, const float* __restrict__ w,
    const float* __restrict__ b, float* __restrict__ out_f32,
    bf16* __restrict__ out_bf16)
{
    const int row = blockIdx.x;
    const int tid = threadIdx.x;
    const float* xr = x + (size_t)row * DMODEL;
    float4 v = ((const float4*)xr)[tid];

    __shared__ float red[256];
    red[tid] = v.x + v.y + v.z + v.w;
    __syncthreads();
    for (int off = 128; off > 0; off >>= 1) {
        if (tid < off) red[tid] += red[tid + off];
        __syncthreads();
    }
    const float mu = red[0] * (1.0f / DMODEL);
    __syncthreads();

    float4 d = {v.x - mu, v.y - mu, v.z - mu, v.w - mu};
    red[tid] = d.x * d.x + d.y * d.y + d.z * d.z + d.w * d.w;
    __syncthreads();
    for (int off = 128; off > 0; off >>= 1) {
        if (tid < off) red[tid] += red[tid + off];
        __syncthreads();
    }
    const float rstd = rsqrtf(red[0] * (1.0f / DMODEL) + 1e-5f);

    const int c = tid * 4;
    float n0 = d.x * rstd * w[c + 0] + b[c + 0];
    float n1 = d.y * rstd * w[c + 1] + b[c + 1];
    float n2 = d.z * rstd * w[c + 2] + b[c + 2];
    float n3 = d.w * rstd * w[c + 3] + b[c + 3];
    float4 o = {n0, n1, n2, n3};
    ((float4*)(out_f32 + (size_t)row * DMODEL))[tid] = o;
    bf16* ob = out_bf16 + (size_t)row * DMODEL + c;
    ob[0] = (bf16)n0; ob[1] = (bf16)n1; ob[2] = (bf16)n2; ob[3] = (bf16)n3;
}

// ---------------------------------------------------------------------------
// 2) WMMA GEMM:  C[M,N] (f32, += residual) = A[M,K] (bf16) * W[K,N] (f32->bf16)
//    64x64 tile per block, 8 waves, each wave -> 16x32 via 2 WMMAs per k-step.
//    Fragment layouts per CDNA5 ISA 7.12.2 (16-bit A 16x32, B 32x16).
// ---------------------------------------------------------------------------
__global__ __launch_bounds__(256) void gemm_bf16_kernel(
    const bf16* __restrict__ A, const float* __restrict__ W,
    const float* __restrict__ residual, float* __restrict__ C,
    int M, int N, int K)
{
    __shared__ __align__(16) bf16 As[64 * 32];   // [m][k] row-major
    __shared__ __align__(32) bf16 Bs[64 * 32];   // [n][k] (W tile transposed)

    const int tid  = threadIdx.x;
    const int lane = tid & 31;
    const int wave = tid >> 5;
    const int mBase = blockIdx.y * 64;
    const int nBase = blockIdx.x * 64;
    const int mw = (wave >> 1) * 16;             // 0,16,32,48
    const int nw = (wave & 1) * 32;              // 0,32

    v8f acc0 = {};
    v8f acc1 = {};

    const int am = tid >> 2;                     // 0..63  (A tile row)
    const int ac = (tid & 3) * 8;                // A tile col chunk
    const int bk = tid >> 3;                     // 0..31  (W tile row = k)
    const int bn = (tid & 7) * 8;                // W tile col chunk

    const int ml = lane & 15;
    const int ka = (lane < 16) ? 0 : 8;          // A frag k-base (split pattern)
    const int kb = (lane < 16) ? 0 : 8;          // B frag k-base (split pattern)

    for (int k0 = 0; k0 < K; k0 += 32) {
        __syncthreads();
        // A tile: 64x32 bf16, one 16B chunk per thread
        uint4 av = *(const uint4*)(A + (size_t)(mBase + am) * K + k0 + ac);
        *(uint4*)(&As[tid * 8]) = av;
        // W tile: 32 rows x 64 cols f32 -> bf16, transposed into Bs[n][k]
        const float* wp = W + (size_t)(k0 + bk) * N + nBase + bn;
        #pragma unroll
        for (int j = 0; j < 8; ++j)
            Bs[(size_t)(bn + j) * 32 + bk] = (bf16)wp[j];
        __syncthreads();

        v8bf alo = *(const v8bf*)(&As[(mw + ml) * 32 + ka]);
        v8bf ahi = *(const v8bf*)(&As[(mw + ml) * 32 + ka + 16]);
        v16bf afrag;
        #pragma unroll
        for (int j = 0; j < 8; ++j) { afrag[j] = alo[j]; afrag[j + 8] = ahi[j]; }

        v8bf b0lo = *(const v8bf*)(&Bs[(nw + ml) * 32 + kb]);
        v8bf b0hi = *(const v8bf*)(&Bs[(nw + ml) * 32 + kb + 16]);
        v8bf b1lo = *(const v8bf*)(&Bs[(nw + 16 + ml) * 32 + kb]);
        v8bf b1hi = *(const v8bf*)(&Bs[(nw + 16 + ml) * 32 + kb + 16]);
        v16bf b0, b1;
        #pragma unroll
        for (int j = 0; j < 8; ++j) { b0[j] = b0lo[j]; b0[j + 8] = b0hi[j]; b1[j] = b1lo[j]; b1[j + 8] = b1hi[j]; }

        acc0 = __builtin_amdgcn_wmma_f32_16x16x32_bf16(
            false, afrag, false, b0, (short)0, acc0, false, false);
        acc1 = __builtin_amdgcn_wmma_f32_16x16x32_bf16(
            false, afrag, false, b1, (short)0, acc1, false, false);
    }

    // C/D layout: VGPR r -> row r (lanes 0-15) / r+8 (lanes 16-31), col = lane&15
    const int colBase = nBase + nw + ml;
    #pragma unroll
    for (int r = 0; r < 8; ++r) {
        const int row = mBase + mw + ((lane < 16) ? r : r + 8);
        const size_t i0 = (size_t)row * N + colBase;
        float v0 = acc0[r], v1 = acc1[r];
        if (residual) { v0 += residual[i0]; v1 += residual[i0 + 16]; }
        C[i0] = v0;
        C[i0 + 16] = v1;
    }
}

// ---------------------------------------------------------------------------
// 3) beta = sigmoid(normed @ Wb);  g = -exp(A_log) * softplus(normed @ Wa + dt)
//    grid = BT blocks, 256 threads; 8 dot-products of length 1024 per row
// ---------------------------------------------------------------------------
__global__ __launch_bounds__(256) void betag_kernel(
    const float* __restrict__ normed, const float* __restrict__ Wb,
    const float* __restrict__ Wa, const float* __restrict__ A_log,
    const float* __restrict__ dt_bias, float* __restrict__ beta,
    float* __restrict__ g)
{
    const int row = blockIdx.x;
    const int tid = threadIdx.x;
    const float* nr = normed + (size_t)row * DMODEL;

    float aB[NH] = {0, 0, 0, 0};
    float aA[NH] = {0, 0, 0, 0};
    #pragma unroll
    for (int j = 0; j < 4; ++j) {
        const int c = tid * 4 + j;
        const float xv = nr[c];
        #pragma unroll
        for (int h = 0; h < NH; ++h) {
            aB[h] += xv * Wb[c * NH + h];
            aA[h] += xv * Wa[c * NH + h];
        }
    }
    __shared__ float redB[NH * 256];
    __shared__ float redA[NH * 256];
    #pragma unroll
    for (int h = 0; h < NH; ++h) { redB[h * 256 + tid] = aB[h]; redA[h * 256 + tid] = aA[h]; }
    __syncthreads();
    for (int off = 128; off > 0; off >>= 1) {
        if (tid < off) {
            #pragma unroll
            for (int h = 0; h < NH; ++h) {
                redB[h * 256 + tid] += redB[h * 256 + tid + off];
                redA[h * 256 + tid] += redA[h * 256 + tid + off];
            }
        }
        __syncthreads();
    }
    if (tid < NH) {
        const int h = tid;
        beta[(size_t)row * NH + h] = sigmoidf_(redB[h * 256]);
        const float a = redA[h * 256] + dt_bias[h];
        const float sp = (a > 20.0f) ? a : log1pf(__expf(a));
        g[(size_t)row * NH + h] = -__expf(A_log[h]) * sp;
    }
}

// ---------------------------------------------------------------------------
// 4) depthwise causal conv(K=4) + SiLU + per-head L2-norm (+scale) for q / k
//    grid = BT*NH blocks, 64 threads, 4 channels/thread (head dim 256)
// ---------------------------------------------------------------------------
__global__ __launch_bounds__(64) void conv_l2_kernel(
    const float* __restrict__ xin, const float* __restrict__ wconv,
    float* __restrict__ out, float scale)
{
    const int idx = blockIdx.x;
    const int h  = idx & (NH - 1);
    const int bt = idx >> 2;
    const int t  = bt & (TSEQ - 1);
    const int b  = bt >> 11;                      // /TSEQ
    const int tid = threadIdx.x;
    const float* base = xin + (size_t)b * TSEQ * DKEY;

    float y[4];
    float ss = 0.0f;
    #pragma unroll
    for (int j = 0; j < 4; ++j) {
        const int c = h * HKD + tid + 64 * j;
        const float* wc = wconv + c * 4;
        float acc = 0.0f;
        #pragma unroll
        for (int i = 0; i < 4; ++i) {
            const int tt = t - 3 + i;
            if (tt >= 0) acc += base[(size_t)tt * DKEY + c] * wc[i];
        }
        y[j] = acc * sigmoidf_(acc);              // SiLU
        ss += y[j] * y[j];
    }
    __shared__ float red[64];
    red[tid] = ss;
    __syncthreads();
    for (int off = 32; off > 0; off >>= 1) {
        if (tid < off) red[tid] += red[tid + off];
        __syncthreads();
    }
    const float rn = rsqrtf(red[0] + 1e-6f) * scale;
    #pragma unroll
    for (int j = 0; j < 4; ++j) {
        const int c = h * HKD + tid + 64 * j;
        out[(size_t)bt * DKEY + c] = y[j] * rn;
    }
}

// ---------------------------------------------------------------------------
// 5) depthwise causal conv + SiLU for v (pure elementwise over B*T*DV)
// ---------------------------------------------------------------------------
__global__ __launch_bounds__(256) void conv_silu_kernel(
    const float* __restrict__ xin, const float* __restrict__ wconv,
    float* __restrict__ out)
{
    const size_t gid = (size_t)blockIdx.x * 256 + threadIdx.x;
    const int c = (int)(gid & (DVAL - 1));
    const size_t bt = gid >> 11;                  // /DVAL
    const int t = (int)(bt & (TSEQ - 1));
    const size_t b = bt >> 11;
    const float* base = xin + b * TSEQ * DVAL;
    const float* wc = wconv + (size_t)c * 4;
    float acc = 0.0f;
    #pragma unroll
    for (int i = 0; i < 4; ++i) {
        const int tt = t - 3 + i;
        if (tt >= 0) acc += base[(size_t)tt * DVAL + c] * wc[i];
    }
    out[gid] = acc * sigmoidf_(acc);
}

// ---------------------------------------------------------------------------
// 6) Gated delta-rule scan. One block per (b, h, v-tile of 64 cols).
//    State slice S[256 x 64] lives in registers: thread (rgrp, vcol) holds
//    S[rgrp*64 .. rgrp*64+63, vcol] (64 VGPRs). Cross-row-group reductions
//    go through LDS. q/k rows prefetched ahead with global_prefetch.
// ---------------------------------------------------------------------------
__global__ __launch_bounds__(256) void scan_kernel(
    const float* __restrict__ q, const float* __restrict__ k,
    const float* __restrict__ v, const float* __restrict__ g,
    const float* __restrict__ beta, float* __restrict__ o)
{
    const int bid = blockIdx.x;
    const int vt = bid & 7;
    const int h  = (bid >> 3) & 3;
    const int b  = bid >> 5;
    const int tid  = threadIdx.x;
    const int vcol = tid & 63;
    const int rg   = (tid >> 6) * 64;             // row-group base: 0,64,128,192

    __shared__ float kS[HKD];
    __shared__ float qS[HKD];
    __shared__ float part[256];
    __shared__ float veff[64];
    __shared__ float scal[2];

    float S[64];
    #pragma unroll
    for (int i = 0; i < 64; ++i) S[i] = 0.0f;

    const float* qp = q + (size_t)b * TSEQ * DKEY + h * HKD;
    const float* kp = k + (size_t)b * TSEQ * DKEY + h * HKD;
    const float* vp = v + (size_t)b * TSEQ * DVAL + h * HVD + vt * 64;
    const float* gp = g + (size_t)b * TSEQ * NH + h;
    const float* bp = beta + (size_t)b * TSEQ * NH + h;
    float* op = o + (size_t)b * TSEQ * DVAL + h * HVD + vt * 64;

    for (int t = 0; t < TSEQ; ++t) {
        __syncthreads();
        kS[tid] = kp[(size_t)t * DKEY + tid];
        qS[tid] = qp[(size_t)t * DKEY + tid];
        if (tid < 64) veff[tid] = vp[(size_t)t * DVAL + tid];   // holds v_t for now
        if (tid == 0) {
            scal[0] = __expf(gp[(size_t)t * NH]);
            scal[1] = bp[(size_t)t * NH];
        }
        if (t + 8 < TSEQ) {                        // hide latency of next rows
            __builtin_prefetch(kp + (size_t)(t + 8) * DKEY + tid, 0, 1);
            __builtin_prefetch(qp + (size_t)(t + 8) * DKEY + tid, 0, 1);
        }
        __syncthreads();

        const float dec = scal[0];
        float pk = 0.0f;
        #pragma unroll
        for (int i = 0; i < 64; ++i) {             // S *= exp(g); k^T S partial
            S[i] *= dec;
            pk += kS[rg + i] * S[i];
        }
        part[tid] = pk;
        __syncthreads();
        if (tid < 64) {                            // v_eff = (v - k^T S) * beta
            const float s = part[tid] + part[tid + 64] + part[tid + 128] + part[tid + 192];
            veff[tid] = (veff[tid] - s) * scal[1];
        }
        __syncthreads();

        const float ve = veff[vcol];
        float pq = 0.0f;
        #pragma unroll
        for (int i = 0; i < 64; ++i) {             // S += k v_eff^T; q^T S partial
            S[i] += kS[rg + i] * ve;
            pq += qS[rg + i] * S[i];
        }
        part[tid] = pq;
        __syncthreads();
        if (tid < 64)
            op[(size_t)t * DVAL + tid] =
                part[tid] + part[tid + 64] + part[tid + 128] + part[tid + 192];
    }
}

// ---------------------------------------------------------------------------
// 7) RMS-norm + swish output gate; emits bf16 for the final WMMA GEMM
//    grid = BT*NH blocks, 128 threads (head dim 512)
// ---------------------------------------------------------------------------
__global__ __launch_bounds__(128) void gate_kernel(
    const float* __restrict__ o, const float* __restrict__ gproj,
    const float* __restrict__ norm_w, bf16* __restrict__ og)
{
    const int idx = blockIdx.x;
    const int h = idx & (NH - 1);
    const size_t bt = (size_t)(idx >> 2);
    const int tid = threadIdx.x;
    const float* orow = o + bt * DVAL + h * HVD;
    const float* grow = gproj + bt * DVAL + h * HVD;
    bf16* out = og + bt * DVAL + h * HVD;

    float vals[4];
    float ss = 0.0f;
    #pragma unroll
    for (int j = 0; j < 4; ++j) {
        const float x = orow[tid + 128 * j];
        vals[j] = x;
        ss += x * x;
    }
    __shared__ float red[128];
    red[tid] = ss;
    __syncthreads();
    for (int off = 64; off > 0; off >>= 1) {
        if (tid < off) red[tid] += red[tid + off];
        __syncthreads();
    }
    const float rms = rsqrtf(red[0] * (1.0f / HVD) + 1e-5f);
    #pragma unroll
    for (int j = 0; j < 4; ++j) {
        const int c = tid + 128 * j;
        const float gv = grow[c];
        float val = vals[j] * rms * norm_w[c];
        val *= gv * sigmoidf_(gv);
        out[c] = (bf16)val;
    }
}

// ---------------------------------------------------------------------------
// Host launch
// ---------------------------------------------------------------------------
extern "C" void kernel_launch(void* const* d_in, const int* in_sizes, int n_in,
                              void* d_out, int out_size, void* d_ws, size_t ws_size,
                              hipStream_t stream)
{
    const float* x       = (const float*)d_in[0];
    const float* ln_w    = (const float*)d_in[1];
    const float* ln_b    = (const float*)d_in[2];
    const float* Wq      = (const float*)d_in[3];
    const float* Wk      = (const float*)d_in[4];
    const float* Wv      = (const float*)d_in[5];
    const float* conv_q  = (const float*)d_in[6];
    const float* conv_k  = (const float*)d_in[7];
    const float* conv_v  = (const float*)d_in[8];
    const float* Wb      = (const float*)d_in[9];
    const float* Wa      = (const float*)d_in[10];
    const float* A_log   = (const float*)d_in[11];
    const float* dt_bias = (const float*)d_in[12];
    const float* Wg      = (const float*)d_in[13];
    const float* norm_w  = (const float*)d_in[14];
    const float* Wo      = (const float*)d_in[15];

    char* ws = (char*)d_ws;
    size_t off = 0;
    auto alloc = [&](size_t bytes) -> char* {
        char* p = ws + off;
        off += (bytes + 255) & ~(size_t)255;
        return p;
    };
    // distinct buffers
    float* normed   = (float*)alloc((size_t)BT * DMODEL * 4);   // 32 MB
    bf16*  normedbf = (bf16*) alloc((size_t)BT * DMODEL * 2);   // 16 MB
    float* qproj    = (float*)alloc((size_t)BT * DKEY * 4);     // 32 MB
    float* kproj    = (float*)alloc((size_t)BT * DKEY * 4);     // 32 MB
    float* vproj    = (float*)alloc((size_t)BT * DVAL * 4);     // 64 MB
    float* gproj    = (float*)alloc((size_t)BT * DVAL * 4);     // 64 MB
    float* vn       = (float*)alloc((size_t)BT * DVAL * 4);     // 64 MB
    float* betaA    = (float*)alloc((size_t)BT * NH * 4);
    float* gA       = (float*)alloc((size_t)BT * NH * 4);
    // aliases (stream is in-order; producer of alias runs after last consumer
    // of the original buffer):
    float* qn   = normed;          // betag (last reader of normed) runs first
    float* kn   = qproj;           // conv-q (reader of qproj) runs first
    float* oA   = vproj;           // conv-v (reader of vproj) runs first
    bf16*  ogbf = (bf16*)kproj;    // conv-k (reader of kproj) runs first

    // 1) LayerNorm
    layernorm_kernel<<<BT, 256, 0, stream>>>(x, ln_w, ln_b, normed, normedbf);
    // 2) WMMA projections
    gemm_bf16_kernel<<<dim3(DKEY / 64, BT / 64), 256, 0, stream>>>(
        normedbf, Wq, nullptr, qproj, BT, DKEY, DMODEL);
    gemm_bf16_kernel<<<dim3(DKEY / 64, BT / 64), 256, 0, stream>>>(
        normedbf, Wk, nullptr, kproj, BT, DKEY, DMODEL);
    gemm_bf16_kernel<<<dim3(DVAL / 64, BT / 64), 256, 0, stream>>>(
        normedbf, Wv, nullptr, vproj, BT, DVAL, DMODEL);
    gemm_bf16_kernel<<<dim3(DVAL / 64, BT / 64), 256, 0, stream>>>(
        normedbf, Wg, nullptr, gproj, BT, DVAL, DMODEL);
    // 3) beta / g (reads normed, must precede the qn alias write)
    betag_kernel<<<BT, 256, 0, stream>>>(normed, Wb, Wa, A_log, dt_bias, betaA, gA);
    // 4) conv + SiLU + l2norm for q (scale HK^-0.5 = 1/16) and k
    conv_l2_kernel<<<BT * NH, 64, 0, stream>>>(qproj, conv_q, qn, 0.0625f);
    conv_l2_kernel<<<BT * NH, 64, 0, stream>>>(kproj, conv_k, kn, 1.0f);
    // 5) conv + SiLU for v
    conv_silu_kernel<<<(BT * DVAL) / 256, 256, 0, stream>>>(vproj, conv_v, vn);
    // 6) gated delta-rule scan (register-resident state, 128 blocks)
    scan_kernel<<<BATCH * NH * 8, 256, 0, stream>>>(qn, kn, vn, gA, betaA, oA);
    // 7) RMS-norm swish gate -> bf16
    gate_kernel<<<BT * NH, 128, 0, stream>>>(oA, gproj, norm_w, ogbf);
    // 8) out = x + o_gated @ Wo  (WMMA, fused residual)
    gemm_bf16_kernel<<<dim3(DMODEL / 64, BT / 64), 256, 0, stream>>>(
        ogbf, Wo, x, (float*)d_out, BT, DMODEL, DVAL);
}
